// EdgeWeightGNN_batch_83545703842236
// MI455X (gfx1250) — matrix operand order repaired and checked by reading, not compiled, not run on hardware
//
#include <hip/hip_runtime.h>
#include <hip/hip_fp16.h>
#include <math.h>

typedef __attribute__((ext_vector_type(16))) _Float16 v16h;
typedef __attribute__((ext_vector_type(8)))  float    v8f;

#define G_GRAPHS 64
#define N_NODES  1024
#define NTOT     65536      // G*N
#define E_DIR    524288     // directed edges in edge_index
#define E_INT    262144     // internal output edges (E_DIR/2)
#define E_OUT    393216     // E_INT + 2*G*N
#define BATCH_OUT 196608    // NTOT + 2*G*N

// ---------------- WMMA fragment loaders (wave32, 16x16x32 f16) ----------------
// A (16x32): lane L<16 holds row L, K = {k0..k0+7, k0+16..k0+23};
//            lane L>=16 holds row L-16, K = {k0+8..k0+15, k0+24..k0+31}.
// Loaded as two aligned 16-byte chunks (callers guarantee 16B alignment).
__device__ inline v16h load_a_frag(const _Float16* rowp, int k0, int hi) {
  const uint4* p1 = (const uint4*)(rowp + k0 + hi * 8);
  const uint4* p2 = (const uint4*)(rowp + k0 + 16 + hi * 8);
  union { uint4 u[2]; v16h v; } t;
  t.u[0] = *p1;
  t.u[1] = *p2;
  return t.v;
}

// B (32x16): weights pre-transposed to [Kout][Kin]; same K order as A:
// lane L<16 -> K = {k0..k0+7, k0+16..k0+23}, lane L>=16 -> K = {k0+8..k0+15, k0+24..k0+31}.
// Two aligned 16-byte reads per lane.
__device__ inline v16h load_b_frag(const _Float16* rowp, int k0, int hi) {
  const uint4* p1 = (const uint4*)(rowp + k0 + hi * 8);
  const uint4* p2 = (const uint4*)(rowp + k0 + 16 + hi * 8);
  union { uint4 u[2]; v16h v; } t;
  t.u[0] = *p1;
  t.u[1] = *p2;
  return t.v;
}

// ---------------- weight prep: fp32 [Kin][Kout] -> f16 transposed [Kout][Kpad]
__global__ void k_transpose_f16(const float* __restrict__ W, __half* __restrict__ Wt,
                                int Kin, int Kout, int Kpad) {
  int i = blockIdx.x * blockDim.x + threadIdx.x;
  int total = Kout * Kpad;
  if (i >= total) return;
  int c = i / Kpad, k = i - c * Kpad;
  float v = (k < Kin) ? W[k * Kout + c] : 0.f;
  ((_Float16*)Wt)[i] = (_Float16)v;
}

// GCN fused weight: rows 0..Din-1 = Wr, Din..2Din-1 = Wk, rest zero.
__global__ void k_prep_gcn_w(const float* __restrict__ Wr, const float* __restrict__ Wk,
                             __half* __restrict__ Wt, int Din, int Dout, int Kpad) {
  int i = blockIdx.x * blockDim.x + threadIdx.x;
  int total = Dout * Kpad;
  if (i >= total) return;
  int c = i / Kpad, k = i - c * Kpad;
  float v = 0.f;
  if (k < Din)          v = Wr[k * Dout + c];
  else if (k < 2 * Din) v = Wk[(k - Din) * Dout + c];
  ((_Float16*)Wt)[i] = (_Float16)v;
}

__global__ void k_zero(float* __restrict__ p, int n) {
  int i = blockIdx.x * blockDim.x + threadIdx.x;
  if (i < n) p[i] = 0.f;
}

// ---------------- GCN scatter: agg[tgt] += h[src] * ew ----------------
__global__ void k_scatter(const float* __restrict__ h, const int* __restrict__ ei,
                          const float* __restrict__ ew, float* __restrict__ agg, int Din) {
  long long i = (long long)blockIdx.x * blockDim.x + threadIdx.x;
  if (i >= (long long)E_DIR * Din) return;
  int e = (int)(i / Din), f = (int)(i - (long long)e * Din);
  int s = ei[e];
  int t = ei[E_DIR + e];
  atomicAdd(&agg[(size_t)t * Din + f], h[(size_t)s * Din + f] * ew[e]);
}

// ---------------- GCN transform: h_out = relu([agg|h] @ Wcat + br), via WMMA --
__global__ __launch_bounds__(32)
void k_gcn_xform(const float* __restrict__ agg, const float* __restrict__ h_in,
                 const __half* __restrict__ Wt_, const float* __restrict__ br,
                 float* __restrict__ h_out, int Din, int Kpad, int Dout) {
  __shared__ __align__(32) _Float16 sAct[16 * 128];
  const _Float16* Wt = (const _Float16*)Wt_;
  int lane = threadIdx.x;
  int base = blockIdx.x * 16;

  for (int idx = lane; idx < 16 * Kpad; idx += 32) {
    int m = idx / Kpad, k = idx - m * Kpad;
    int node = base + m;
    float v = 0.f;
    if (k < Din)          v = agg[(size_t)node * Din + k];
    else if (k < 2 * Din) v = h_in[(size_t)node * Din + (k - Din)];
    sAct[m * Kpad + k] = (_Float16)v;
  }
  __syncthreads();

  int c = lane & 15, hi = lane >> 4;
  for (int n0 = 0; n0 < Dout; n0 += 16) {
    v8f acc = {};
    for (int k0 = 0; k0 < Kpad; k0 += 32) {
      v16h a = load_a_frag(sAct + c * Kpad, k0, hi);
      v16h b = load_b_frag(Wt + (size_t)(n0 + c) * Kpad, k0, hi);
      acc = __builtin_amdgcn_wmma_f32_16x16x32_f16(false, a, false, b,
                                                   (short)0, acc, false, false);
    }
    float bv = br[n0 + c];
    int mb = hi * 8;
#pragma unroll
    for (int i = 0; i < 8; ++i) {
      float v = acc[i] + bv;
      v = v > 0.f ? v : 0.f;
      h_out[(size_t)(base + mb + i) * Dout + n0 + c] = v;
    }
  }
}

// ---------------- edge MLP: 256->256->128->64->32 (WMMA) -> 1 (sigmoid) ------
// mode 0: internal edges (gather via edge_index). mode 1: boundary-left (+h,+h).
// mode 2: boundary-right (+h,-h).
__global__ __launch_bounds__(32)
void k_mlp(const float* __restrict__ h, const int* __restrict__ ei,
           const __half* __restrict__ Wt_,
           const float* __restrict__ b0, const float* __restrict__ b1,
           const float* __restrict__ b2, const float* __restrict__ b3,
           const float* __restrict__ W4, const float* __restrict__ b4,
           float* __restrict__ ew_out, int mode) {
  __shared__ __align__(32) _Float16 sA[16 * 256];
  __shared__ __align__(32) _Float16 sB[16 * 256];
  const _Float16* Wt = (const _Float16*)Wt_;
  int lane = threadIdx.x;
  int tile = blockIdx.x;

  // gather 16 edge embeddings (16 x 256) as f16 into LDS
  for (int idx = lane; idx < 16 * 256; idx += 32) {
    int m = idx >> 8, k = idx & 255;
    int item = tile * 16 + m;
    int rnode, cnode;
    float sgn;
    if (mode == 0) { rnode = ei[item]; cnode = ei[E_DIR + item]; sgn = 1.f; }
    else           { rnode = item;     cnode = item; sgn = (mode == 2) ? -1.f : 1.f; }
    float v = (k < 128) ? h[(size_t)rnode * 128 + k]
                        : sgn * h[(size_t)cnode * 128 + (k - 128)];
    sA[idx] = (_Float16)v;
  }
  __syncthreads();

  const int   KIN[4]  = {256, 256, 128, 64};
  const int   KOUT[4] = {256, 128, 64, 32};
  const int   WOFF[4] = {0, 65536, 98304, 106496};
  const float* BIAS[4] = {b0, b1, b2, b3};

  _Float16* bufs[2] = {sA, sB};
  int cur = 0;
  int c = lane & 15, hi = lane >> 4;

  for (int L = 0; L < 4; ++L) {
    const _Float16* in = bufs[cur];
    _Float16* out = bufs[cur ^ 1];
    int kin = KIN[L], kout = KOUT[L];
    const _Float16* w = Wt + WOFF[L];
    for (int n0 = 0; n0 < kout; n0 += 16) {
      v8f acc = {};
      for (int k0 = 0; k0 < kin; k0 += 32) {
        v16h a = load_a_frag(in + c * 256, k0, hi);
        v16h b = load_b_frag(w + (size_t)(n0 + c) * kin, k0, hi);
        acc = __builtin_amdgcn_wmma_f32_16x16x32_f16(false, a, false, b,
                                                     (short)0, acc, false, false);
      }
      float bv = BIAS[L][n0 + c];
      int mb = hi * 8;
#pragma unroll
      for (int i = 0; i < 8; ++i) {
        float v = acc[i] + bv;
        v = v > 0.f ? v : 0.f;
        out[(mb + i) * 256 + n0 + c] = (_Float16)v;
      }
    }
    __syncthreads();
    cur ^= 1;
  }

  // final 32->1 + sigmoid, one lane per row (fp32 accumulation)
  const _Float16* fin = bufs[cur];
  if (lane < 16) {
    float s = b4[0];
#pragma unroll
    for (int k = 0; k < 32; ++k) s += (float)fin[lane * 256 + k] * W4[k];
    float o = 1.f / (1.f + __expf(-s));
    int item = tile * 16 + lane;
    size_t pos;
    if (mode == 0) pos = (size_t)item;
    else {
      int g = item >> 10, j = item & 1023;
      pos = (size_t)E_INT + (size_t)g * 2048 + j + (mode == 2 ? 1024 : 0);
    }
    ew_out[pos] = o;
  }
}

// ---------------- output index / batch reconstruction ----------------
__global__ void k_edge_idx(const int* __restrict__ ei,
                           int* __restrict__ out0, int* __restrict__ out1) {
  int i = blockIdx.x * blockDim.x + threadIdx.x;
  if (i >= E_OUT) return;
  int r0, r1;
  if (i < E_INT) { r0 = ei[i]; r1 = ei[E_DIR + i]; }
  else {
    int k = i - E_INT;
    int g = k >> 11;
    int rem = k & 2047;
    int t = rem >> 10;
    int j = rem & 1023;
    r0 = g * N_NODES + j;                       // real node
    r1 = NTOT + g * 2048 + j + (t ? 1024 : 0);  // boundary node
  }
  out0[i] = r0;
  out1[i] = r1;
}

__global__ void k_batch_out(const int* __restrict__ batch, int* __restrict__ out) {
  int i = blockIdx.x * blockDim.x + threadIdx.x;
  if (i >= BATCH_OUT) return;
  out[i] = (i < NTOT) ? batch[i] : ((i - NTOT) >> 11);
}

// ---------------- host orchestration ----------------
static inline int cdiv(long long a, int b) { return (int)((a + b - 1) / b); }

extern "C" void kernel_launch(void* const* d_in, const int* in_sizes, int n_in,
                              void* d_out, int out_size, void* d_ws, size_t ws_size,
                              hipStream_t stream) {
  const float* x     = (const float*)d_in[0];
  const int*   ei    = (const int*)d_in[1];
  const float* ew    = (const float*)d_in[2];
  const int*   batch = (const int*)d_in[3];

  const float* Wr[3] = {(const float*)d_in[5], (const float*)d_in[8],  (const float*)d_in[11]};
  const float* br[3] = {(const float*)d_in[6], (const float*)d_in[9],  (const float*)d_in[12]};
  const float* Wk[3] = {(const float*)d_in[7], (const float*)d_in[10], (const float*)d_in[13]};
  const int mbase[3] = {14, 24, 34};  // mlp_internal, mlp_left, mlp_right

  // workspace layout (all sub-buffers 32B-aligned)
  char*  ws   = (char*)d_ws;
  float* bufA = (float*)(ws);                        // 65536*128 f32 = 32 MiB
  float* bufB = (float*)(ws + (size_t)33554432);     // 32 MiB
  float* agg  = (float*)(ws + (size_t)67108864);     // 65536*64 f32 = 16 MiB
  __half* wg[3];
  wg[0] = (__half*)(ws + (size_t)83886080);          // 32*32
  wg[1] = wg[0] + 1024;                              // 64*64
  wg[2] = wg[1] + 4096;                              // 128*128
  __half* wm[3];
  wm[0] = wg[2] + 16384;                             // 108544 halves each
  wm[1] = wm[0] + 108544;
  wm[2] = wm[1] + 108544;

  // --- stage weights as f16 (transposed [Kout][Kpad]) ---
  const int gDin[3]  = {4, 32, 64};
  const int gKpad[3] = {32, 64, 128};
  const int gDout[3] = {32, 64, 128};
  for (int l = 0; l < 3; ++l) {
    int tot = gDout[l] * gKpad[l];
    k_prep_gcn_w<<<cdiv(tot, 256), 256, 0, stream>>>(Wr[l], Wk[l], wg[l],
                                                     gDin[l], gDout[l], gKpad[l]);
  }
  const int mKin[4]  = {256, 256, 128, 64};
  const int mKout[4] = {256, 128, 64, 32};
  const int mWoff[4] = {0, 65536, 98304, 106496};
  for (int m = 0; m < 3; ++m) {
    for (int L = 0; L < 4; ++L) {
      const float* W = (const float*)d_in[mbase[m] + 2 * L];
      int tot = mKout[L] * mKin[L];
      k_transpose_f16<<<cdiv(tot, 256), 256, 0, stream>>>(W, wm[m] + mWoff[L],
                                                          mKin[L], mKout[L], mKin[L]);
    }
  }

  // --- GCN layers ---
  const float* h_in[3]  = {x, bufA, bufB};
  float*       h_out[3] = {bufA, bufB, bufA};
  for (int l = 0; l < 3; ++l) {
    int nz = NTOT * gDin[l];
    k_zero<<<cdiv(nz, 256), 256, 0, stream>>>(agg, nz);
    long long sw = (long long)E_DIR * gDin[l];
    k_scatter<<<cdiv(sw, 256), 256, 0, stream>>>(h_in[l], ei, ew, agg, gDin[l]);
    k_gcn_xform<<<NTOT / 16, 32, 0, stream>>>(agg, h_in[l], wg[l], br[l], h_out[l],
                                              gDin[l], gKpad[l], gDout[l]);
  }
  const float* h3 = bufA;  // 65536 x 128

  // --- outputs layout ---
  int*   out_idx0  = (int*)d_out;
  int*   out_idx1  = out_idx0 + E_OUT;
  float* out_ew    = (float*)d_out + 2 * (size_t)E_OUT;
  int*   out_batch = (int*)d_out + 3 * (size_t)E_OUT;

  // --- MLPs over edge tiles ---
  for (int m = 0; m < 3; ++m) {
    int mb = mbase[m];
    int tiles = (m == 0) ? (E_INT / 16) : (NTOT / 16);
    k_mlp<<<tiles, 32, 0, stream>>>(h3, ei, wm[m],
                                    (const float*)d_in[mb + 1], (const float*)d_in[mb + 3],
                                    (const float*)d_in[mb + 5], (const float*)d_in[mb + 7],
                                    (const float*)d_in[mb + 8], (const float*)d_in[mb + 9],
                                    out_ew, m);
  }

  // --- edge_index_out + batch_out ---
  k_edge_idx<<<cdiv(E_OUT, 256), 256, 0, stream>>>(ei, out_idx0, out_idx1);
  k_batch_out<<<cdiv(BATCH_OUT, 256), 256, 0, stream>>>(batch, out_batch);
}
